// BachelardianImaginationSpace_27221502722617
// MI455X (gfx1250) — compile-verified, matrix operand order checked by reading, not run on hardware
//
#include <hip/hip_runtime.h>

#define VOCAB 32000
#define DIM   512
#define HID   1024
#define BB    16
#define TT    256
#define TCH   4        // timesteps per logits block (B-tile reuse factor)

typedef __attribute__((ext_vector_type(16))) __bf16 v16bf;
typedef __attribute__((ext_vector_type(8)))  __bf16 v8bf;
typedef __attribute__((ext_vector_type(8)))  float  v8f;

// D = A(16x32 bf16) * B(32x16 bf16) + C(16x16 f32)
__device__ __forceinline__ v8f wmma_bf16(v16bf a, v16bf b, v8f c) {
    return __builtin_amdgcn_wmma_f32_16x16x32_bf16(
        /*neg_a=*/false, a, /*neg_b=*/false, b,
        /*c_mod=*/(short)0, c, /*reuse_a=*/false, /*reuse_b=*/false);
}

// ---- tile loaders ------------------------------------------------------
// ISA 16-bit A/B layout for 16x16x32: lane L, row/col = L%16, kgrp = L/16.
// elements 0..7  -> K = k0 + kgrp*8 + j
// elements 8..15 -> K = k0 + 16 + kgrp*8 + (j-8)

// bf16 row (LDS or global) -> two 16B vector loads, register concat
__device__ __forceinline__ v16bf load16_bf(const __bf16* __restrict__ row,
                                           int k0, int kgrp) {
    v8bf lo = *(const v8bf*)(row + k0 + kgrp * 8);
    v8bf hi = *(const v8bf*)(row + k0 + 16 + kgrp * 8);
    return __builtin_shufflevector(lo, hi, 0, 1, 2, 3, 4, 5, 6, 7,
                                   8, 9, 10, 11, 12, 13, 14, 15);
}

// fp32 row (global), converting to bf16, optional sign (fallback path)
__device__ __forceinline__ v16bf load16_f32(const float* __restrict__ row,
                                            int k0, int kgrp, float sgn) {
    v16bf r;
    const float* p0 = row + k0 + kgrp * 8;
    const float* p1 = row + k0 + 16 + kgrp * 8;
#pragma unroll
    for (int j = 0; j < 8; ++j) r[j] = (__bf16)(sgn * p0[j]);
#pragma unroll
    for (int j = 0; j < 8; ++j) r[8 + j] = (__bf16)(sgn * p1[j]);
    return r;
}

template <bool PRE>
__device__ __forceinline__ v16bf loadAB(const float* rowf, const __bf16* rowb,
                                        int k0, int kgrp, float sgn) {
    if constexpr (PRE) return load16_bf(rowb, k0, kgrp);
    else               return load16_f32(rowf, k0, kgrp, sgn);
}

__device__ __forceinline__ float sigmoidf_(float x) {
    return 1.0f / (1.0f + __expf(-x));
}

// ---- phase 0: fp32 -> bf16 (vectorized, optional sign fold) ------------
__global__ __launch_bounds__(256)
void cvt_f32_to_bf16(const float* __restrict__ src, __bf16* __restrict__ dst,
                     long n8, float sgn) {
    long i = (long)blockIdx.x * blockDim.x + threadIdx.x; // 8-element group
    if (i >= n8) return;
    const float4* s = (const float4*)src + 2 * i;
    float4 a = s[0], b = s[1];
    v8bf d;
    d[0] = (__bf16)(sgn * a.x); d[1] = (__bf16)(sgn * a.y);
    d[2] = (__bf16)(sgn * a.z); d[3] = (__bf16)(sgn * a.w);
    d[4] = (__bf16)(sgn * b.x); d[5] = (__bf16)(sgn * b.y);
    d[6] = (__bf16)(sgn * b.z); d[7] = (__bf16)(sgn * b.w);
    *((v8bf*)dst + i) = d;
}

// ---- phase 1: sequential GRU recurrence (one WGP, 32 waves) ------------
template <bool PRE>
__global__ __launch_bounds__(1024, 1)
void gru_recurrence(const int*    __restrict__ tokens,
                    const float*  __restrict__ emb_f,
                    const __bf16* __restrict__ emb_b,
                    const float*  __restrict__ atticW_f,
                    const __bf16* __restrict__ atticW_b,
                    const float*  __restrict__ atticB,
                    const float*  __restrict__ cellarW_f,
                    const __bf16* __restrict__ cellarW_b, // pre-negated
                    const float*  __restrict__ cellarB,
                    const float*  __restrict__ W_ih_f,
                    const __bf16* __restrict__ W_ih_b,
                    const float*  __restrict__ b_ih,
                    const float*  __restrict__ W_hh_f,
                    const __bf16* __restrict__ W_hh_b,
                    const float*  __restrict__ b_hh,
                    __bf16*       __restrict__ h_hist,   // [T,B,HID] bf16
                    float*        __restrict__ h_final)  // [B,HID] f32
{
    __shared__ __bf16 sX[BB][2 * HID]; // 64 KB: concat(attic, cellar)
    __shared__ __bf16 sH[BB][HID];     // 32 KB: hidden state
    __shared__ __bf16 sR[BB][HID];     // 32 KB: r gate
    __shared__ __bf16 sZ[BB][HID];     // 32 KB: z gate   (total 160 KB)

    const int lane = threadIdx.x & 31;
    const int wave = threadIdx.x >> 5;   // 0..31
    const int col  = lane & 15;          // A-row / B-col / D-col
    const int kgrp = lane >> 4;          // K half select

    for (int i = threadIdx.x; i < BB * HID; i += blockDim.x)
        (&sH[0][0])[i] = (__bf16)0.0f;
    __syncthreads();

    for (int t = 0; t < TT; ++t) {
        const int tok = tokens[col * TT + t];             // tokens[B][T]
        const float*  embrow_f = emb_f + (long)tok * DIM;
        const __bf16* embrow_b = emb_b + (long)tok * DIM;

        // ---- Pass A: attic/cellar -> sX -------------------------------
        for (int tile = wave; tile < 128; tile += 32) {
            const bool cel = tile >= 64;
            const int  n0  = (tile & 63) * 16;
            const float*  Wf = cel ? cellarW_f : atticW_f;
            const __bf16* Wb = cel ? cellarW_b : atticW_b;
            const float*  Bv = cel ? cellarB : atticB;
            const float   sg = cel ? -1.0f : 1.0f; // A-side sign (fallback only)
            const float*  wrowf = Wf + (long)(n0 + col) * DIM;
            const __bf16* wrowb = Wb + (long)(n0 + col) * DIM;
            v8f acc = {};
#pragma unroll 4
            for (int k0 = 0; k0 < DIM; k0 += 32) {
                v16bf a = loadAB<PRE>(embrow_f, embrow_b, k0, kgrp,
                                      PRE ? 1.0f : sg);
                v16bf b = loadAB<PRE>(wrowf, wrowb, k0, kgrp, 1.0f);
                acc = wmma_bf16(a, b, acc);
            }
            const float bv   = Bv[n0 + col];
            const int   xcol = (cel ? HID : 0) + n0 + col;
#pragma unroll
            for (int v = 0; v < 8; ++v) {
                int m = v + 8 * kgrp;
                sX[m][xcol] = (__bf16)tanhf(acc[v] + bv);
            }
        }
        __syncthreads();

        // ---- Pass B: fused gi+gh for r,z gates ------------------------
        for (int tile = wave; tile < 128; tile += 32) {
            const int g0 = tile * 16;  // gate col 0..2047
            const float*  wif = W_ih_f + (long)(g0 + col) * (2 * HID);
            const __bf16* wib = W_ih_b + (long)(g0 + col) * (2 * HID);
            const float*  whf = W_hh_f + (long)(g0 + col) * HID;
            const __bf16* whb = W_hh_b + (long)(g0 + col) * HID;
            v8f acc = {};
#pragma unroll 4
            for (int k0 = 0; k0 < 2 * HID; k0 += 32) {
                v16bf a = load16_bf(sX[col], k0, kgrp);
                v16bf b = loadAB<PRE>(wif, wib, k0, kgrp, 1.0f);
                acc = wmma_bf16(a, b, acc);
            }
#pragma unroll 4
            for (int k0 = 0; k0 < HID; k0 += 32) {
                v16bf a = load16_bf(sH[col], k0, kgrp);
                v16bf b = loadAB<PRE>(whf, whb, k0, kgrp, 1.0f);
                acc = wmma_bf16(a, b, acc);
            }
            const float bv  = b_ih[g0 + col] + b_hh[g0 + col];
            const bool  isR = (g0 < HID);
#pragma unroll
            for (int v = 0; v < 8; ++v) {
                int m = v + 8 * kgrp;
                float g = sigmoidf_(acc[v] + bv);
                if (isR) sR[m][g0 + col]       = (__bf16)g;
                else     sZ[m][g0 + col - HID] = (__bf16)g;
            }
        }
        __syncthreads();

        // ---- Pass C: n gate (gi_n + r*gh_n) and GRU update ------------
        v8f accI[2], accH[2];
#pragma unroll
        for (int i = 0; i < 2; ++i) {
            const int tile = wave + 32 * i;               // 0..63
            const int g0   = 2 * HID + tile * 16;
            const float*  wif = W_ih_f + (long)(g0 + col) * (2 * HID);
            const __bf16* wib = W_ih_b + (long)(g0 + col) * (2 * HID);
            const float*  whf = W_hh_f + (long)(g0 + col) * HID;
            const __bf16* whb = W_hh_b + (long)(g0 + col) * HID;
            accI[i] = (v8f){};
            accH[i] = (v8f){};
#pragma unroll 4
            for (int k0 = 0; k0 < 2 * HID; k0 += 32) {
                v16bf a = load16_bf(sX[col], k0, kgrp);
                v16bf b = loadAB<PRE>(wif, wib, k0, kgrp, 1.0f);
                accI[i] = wmma_bf16(a, b, accI[i]);
            }
#pragma unroll 4
            for (int k0 = 0; k0 < HID; k0 += 32) {
                v16bf a = load16_bf(sH[col], k0, kgrp);
                v16bf b = loadAB<PRE>(whf, whb, k0, kgrp, 1.0f);
                accH[i] = wmma_bf16(a, b, accH[i]);
            }
        }
        __syncthreads();   // all GEMM reads of sH/sX done before h update
#pragma unroll
        for (int i = 0; i < 2; ++i) {
            const int tile = wave + 32 * i;
            const int g0   = 2 * HID + tile * 16;
            const int hc   = tile * 16 + col;
            const float bi = b_ih[g0 + col];
            const float bh = b_hh[g0 + col];
#pragma unroll
            for (int v = 0; v < 8; ++v) {
                int m = v + 8 * kgrp;
                float rr   = (float)sR[m][hc];
                float zz   = (float)sZ[m][hc];
                float nn   = tanhf(accI[i][v] + bi + rr * (accH[i][v] + bh));
                float hold = (float)sH[m][hc];
                float hnew = (1.0f - zz) * nn + zz * hold;
                __bf16 hb  = (__bf16)hnew;
                sH[m][hc] = hb;
                h_hist[((long)t * BB + m) * HID + hc] = hb;
                if (t == TT - 1) h_final[m * HID + hc] = hnew;
            }
        }
        __syncthreads();
    }
}

// ---- phase 2: logits GEMM over (t-chunk, vocab tiles) ------------------
// Each block: TCH consecutive timesteps x 8 vocab tiles (one per wave).
// B tile is loaded once per K-step and reused by TCH WMMAs -> 4x less L2
// weight traffic; inner loop is 4 v_wmma per 2 global_load_b128.
template <bool PRE>
__global__ __launch_bounds__(256)
void logits_kernel(const __bf16* __restrict__ h_hist,
                   const float*  __restrict__ outW_f,
                   const __bf16* __restrict__ outW_b,
                   const float*  __restrict__ out_b,
                   float*        __restrict__ out)   // [B,T,VOCAB]
{
    __shared__ __bf16 sA[TCH][BB][HID];   // 128 KB: h_{t0..t0+3}
    const int lane = threadIdx.x & 31;
    const int wave = threadIdx.x >> 5;   // 0..7
    const int col  = lane & 15;
    const int kgrp = lane >> 4;
    const int t0   = blockIdx.y * TCH;

    { // cooperative 128-bit staging: TCH consecutive h_t are contiguous
        const uint4* s = (const uint4*)(h_hist + (long)t0 * BB * HID);
        uint4* d = (uint4*)(&sA[0][0][0]);
        for (int i = threadIdx.x; i < (TCH * BB * HID * 2) / 16; i += blockDim.x)
            d[i] = s[i];
    }
    __syncthreads();

    const int vtile = blockIdx.x * 8 + wave;   // 0..1999
    const int n0    = vtile * 16;
    const float*  wrowf = outW_f + (long)(n0 + col) * HID;
    const __bf16* wrowb = outW_b + (long)(n0 + col) * HID;
    v8f acc[TCH];
#pragma unroll
    for (int u = 0; u < TCH; ++u) acc[u] = (v8f){};

#pragma unroll 2
    for (int k0 = 0; k0 < HID; k0 += 32) {
        v16bf b = loadAB<PRE>(wrowf, wrowb, k0, kgrp, 1.0f);
#pragma unroll
        for (int u = 0; u < TCH; ++u) {
            v16bf a = load16_bf(sA[u][col], k0, kgrp);
            acc[u] = wmma_bf16(a, b, acc[u]);
        }
    }
    const float bv = out_b[n0 + col];
#pragma unroll
    for (int u = 0; u < TCH; ++u) {
#pragma unroll
        for (int v = 0; v < 8; ++v) {
            int m = v + 8 * kgrp;
            out[((long)m * TT + (t0 + u)) * VOCAB + n0 + col] = acc[u][v] + bv;
        }
    }
}

extern "C" void kernel_launch(void* const* d_in, const int* in_sizes, int n_in,
                              void* d_out, int out_size, void* d_ws, size_t ws_size,
                              hipStream_t stream) {
    const int*   tokens  = (const int*)  d_in[0];
    const float* emb     = (const float*)d_in[1];
    const float* atticW  = (const float*)d_in[2];
    const float* atticB  = (const float*)d_in[3];
    const float* cellarW = (const float*)d_in[4];
    const float* cellarB = (const float*)d_in[5];
    const float* W_ih    = (const float*)d_in[6];
    const float* b_ih    = (const float*)d_in[7];
    const float* W_hh    = (const float*)d_in[8];
    const float* b_hh    = (const float*)d_in[9];
    const float* out_W   = (const float*)d_in[10];
    const float* out_b   = (const float*)d_in[11];

    float* out     = (float*)d_out;
    float* h_final = out + (long)BB * TT * VOCAB;

    // workspace layout (bf16 element counts)
    const long N_HHIST  = (long)TT * BB * HID;     //  4.19 M
    const long N_EMB    = (long)VOCAB * DIM;       // 16.38 M
    const long N_ATTIC  = (long)HID * DIM;         //  0.52 M
    const long N_CELLAR = (long)HID * DIM;
    const long N_WIH    = (long)3 * HID * 2 * HID; //  6.29 M
    const long N_WHH    = (long)3 * HID * HID;     //  3.15 M
    const long N_OUTW   = (long)VOCAB * HID;       // 32.77 M

    __bf16* p       = (__bf16*)d_ws;
    __bf16* h_hist  = p;            p += N_HHIST;
    __bf16* emb_b   = p;            p += N_EMB;
    __bf16* attic_b = p;            p += N_ATTIC;
    __bf16* cell_b  = p;            p += N_CELLAR;
    __bf16* wih_b   = p;            p += N_WIH;
    __bf16* whh_b   = p;            p += N_WHH;
    __bf16* outw_b  = p;            p += N_OUTW;

    const size_t need = (size_t)(p - (__bf16*)d_ws) * sizeof(__bf16); // ~128 MB
    const bool   pre  = (ws_size >= need);

    dim3 lgrid(VOCAB / (16 * 8), TT / TCH);   // (250, 64)

    if (pre) {
        auto cvt = [&](const float* s, __bf16* d, long n, float sgn) {
            long n8 = n / 8;
            cvt_f32_to_bf16<<<(unsigned)((n8 + 255) / 256), 256, 0, stream>>>(
                s, d, n8, sgn);
        };
        cvt(emb,     emb_b,   N_EMB,    1.0f);
        cvt(atticW,  attic_b, N_ATTIC,  1.0f);
        cvt(cellarW, cell_b,  N_CELLAR, -1.0f); // fold (-emb) into weights
        cvt(W_ih,    wih_b,   N_WIH,    1.0f);
        cvt(W_hh,    whh_b,   N_WHH,    1.0f);
        cvt(out_W,   outw_b,  N_OUTW,   1.0f);

        gru_recurrence<true><<<1, 1024, 0, stream>>>(
            tokens, emb, emb_b, atticW, attic_b, atticB,
            cellarW, cell_b, cellarB, W_ih, wih_b, b_ih,
            W_hh, whh_b, b_hh, h_hist, h_final);

        logits_kernel<true><<<lgrid, 256, 0, stream>>>(
            h_hist, out_W, outw_b, out_b, out);
    } else {
        gru_recurrence<false><<<1, 1024, 0, stream>>>(
            tokens, emb, emb_b, atticW, attic_b, atticB,
            cellarW, cell_b, cellarB, W_ih, wih_b, b_ih,
            W_hh, whh_b, b_hh, h_hist, h_final);

        logits_kernel<false><<<lgrid, 256, 0, stream>>>(
            h_hist, out_W, outw_b, out_b, out);
    }
}
